// CRF_84679575208548
// MI455X (gfx1250) — matrix operand order repaired and checked by reading, not compiled, not run on hardware
//
#include <hip/hip_runtime.h>
#include <hip/hip_bf16.h>

#define BB 256
#define LL 1024
#define TT 128

typedef __attribute__((ext_vector_type(16))) __bf16 v16bf;
typedef __attribute__((ext_vector_type(8)))  float  v8f;

union V16BF { v16bf v; unsigned short s[16]; };

__device__ __forceinline__ unsigned short f2bf(float f) {
  unsigned int u = __float_as_uint(f);
  u += 0x7FFFu + ((u >> 16) & 1u);      // round-to-nearest-even
  return (unsigned short)(u >> 16);
}

// Raw transcendentals: args here are never denormal-problematic, so skip the
// libm denorm-guard wrappers (v_cmp/v_cndmask/v_ldexp per call).
__device__ __forceinline__ float fast_exp(float x) {   // exp(x), x <= ~0
#if __has_builtin(__builtin_amdgcn_exp2f)
  return __builtin_amdgcn_exp2f(x * 1.4426950408889634f);
#else
  return __expf(x);
#endif
}
__device__ __forceinline__ float fast_log(float x) {   // log(x), x >= ~0.2
#if __has_builtin(__builtin_amdgcn_logf)
  return __builtin_amdgcn_logf(x) * 0.6931471805599453f;
#else
  return __logf(x);
#endif
}

// ---------------------------------------------------------------------------
// Kernel A: gold-path score (unary + binary) per sequence -> ws[b]
// ---------------------------------------------------------------------------
__global__ void __launch_bounds__(256)
crf_score_kernel(const float* __restrict__ inputs,
                 const int*   __restrict__ tags,
                 const int*   __restrict__ lens,
                 const float* __restrict__ trans,
                 float*       __restrict__ score_ws)
{
  const int b   = blockIdx.x;
  const int tid = threadIdx.x;
  const int len = lens[b];
  const int*   tg  = tags   + (size_t)b * LL;
  const float* inp = inputs + (size_t)b * LL * TT;

  float acc = 0.f;
  for (int t = tid; t < len; t += 256) {
    const int tag = tg[t];
    acc += inp[(size_t)t * TT + tag];               // unary, mask t < len
    if (t >= 1) acc += trans[tg[t - 1] * TT + tag]; // binary, mask[:,1:]
  }
  __shared__ float red[256];
  red[tid] = acc;
  __syncthreads();
  for (int s = 128; s > 0; s >>= 1) {
    if (tid < s) red[tid] += red[tid + s];
    __syncthreads();
  }
  if (tid == 0) score_ws[b] = red[0];
}

// ---------------------------------------------------------------------------
// Kernel B: forward algorithm. Each block owns a 16-row batch tile.
// Per step: alpha' = emission + m + log( exp(alpha-m) @ exp(trans) )  via WMMA.
// 2 barriers/step; row reductions via intra-wave shuffles (2 rows per wave32).
// ---------------------------------------------------------------------------
__global__ void __launch_bounds__(256)
crf_forward_kernel(const float* __restrict__ inputs,
                   const int*   __restrict__ lens,
                   const float* __restrict__ trans,
                   const float* __restrict__ score_ws,
                   float*       __restrict__ out)
{
  const int tid  = threadIdx.x;
  const int lane = tid & 31;
  const int wave = tid >> 5;          // 0..7: N-tile [16w, 16w+16)
  const int hi   = (lane >> 4) & 1;   // half-wave select
  const int lm16 = lane & 15;
  const int row  = tid >> 4;          // 0..15: reduction role (contiguous in-wave)
  const int sub  = tid & 15;
  const int b0   = blockIdx.x * 16;

  __shared__ float          s_alpha[16][TT + 4];
  __shared__ unsigned short s_ea[16][TT + 8];     // exp(alpha - rowmax) in bf16
  __shared__ float          s_rowmax[16];
  __shared__ float          s_lognorm[16];
  __shared__ int            s_snap[16];
  __shared__ int            s_tmax;

  // snapshot time per row: clip(len-1, 1, L-1)  (== reference's alphas index + 1)
  if (tid < 16) {
    int sn = lens[b0 + tid] - 1;
    sn = sn < 1 ? 1 : sn;
    sn = sn > (LL - 1) ? (LL - 1) : sn;
    s_snap[tid] = sn;
  }
  // init alpha = inputs[:, 0, :]
  #pragma unroll
  for (int i = 0; i < 8; ++i) {
    const int c = sub + 16 * i;
    s_alpha[row][c] = inputs[(size_t)(b0 + row) * LL * TT + c];
  }
  __syncthreads();
  if (tid == 0) {
    int tm = 1;
    #pragma unroll
    for (int i = 0; i < 16; ++i) tm = max(tm, s_snap[i]);
    s_tmax = tm;
  }
  __syncthreads();
  const int tmax = s_tmax;

  // Preload constant B fragments: exp(trans) in bf16, kept in VGPRs all loop.
  // B 32x16 in this kernel's contiguous k order (same as A): lanes 0-15 hold K=0..15, lanes 16-31 K=16..31.
  V16BF bf[4];
  #pragma unroll
  for (int kc = 0; kc < 4; ++kc) {
    #pragma unroll
    for (int e = 0; e < 16; ++e) {
      const int kk = 32 * kc + e + hi * 16;
      const int n  = 16 * wave + lm16;
      bf[kc].s[e] = f2bf(fast_exp(trans[kk * TT + n]));
    }
  }

  const int n = 16 * wave + lm16;     // this lane's output column

  for (int t = 1; t <= tmax; ++t) {
    // ---- issue emission loads for this step up front (overlap with phase 1) ----
    float em[8];
    #pragma unroll
    for (int r = 0; r < 8; ++r) {
      const int m = r + hi * 8;
      em[r] = inputs[((size_t)(b0 + m) * LL + t) * TT + n];
    }

    // ---- phase 1: row max via intra-wave butterfly (16-lane groups) ----
    float lm = -3.0e38f;
    #pragma unroll
    for (int i = 0; i < 8; ++i) lm = fmaxf(lm, s_alpha[row][sub + 16 * i]);
    #pragma unroll
    for (int off = 8; off > 0; off >>= 1) lm = fmaxf(lm, __shfl_xor(lm, off, 16));
    if (sub == 0) s_rowmax[row] = lm;

    // ---- phase 1b: exp(alpha - m) -> bf16 LDS, row-sum as by-product ----
    float ls = 0.f;
    #pragma unroll
    for (int i = 0; i < 8; ++i) {
      const int c = sub + 16 * i;
      const float e = fast_exp(s_alpha[row][c] - lm);
      s_ea[row][c] = f2bf(e);
      ls += e;
    }
    #pragma unroll
    for (int off = 8; off > 0; off >>= 1) ls += __shfl_xor(ls, off, 16);
    // free log_norm snapshot: current alpha is alpha after step (t-1)
    if (sub == 0 && s_snap[row] == t - 1) s_lognorm[row] = lm + fast_log(ls);
    __syncthreads();   // s_ea + s_rowmax visible to all waves

    // ---- phase 2: WMMA  exp(alpha-m) (16x128) @ exp(trans) (128x16-tile) ----
    // A 16x32 in contiguous k order: lanes 0-15 K=0..15, lanes 16-31 K=16..31 (matches B)
    V16BF af[4];
    #pragma unroll
    for (int kc = 0; kc < 4; ++kc) {
      #pragma unroll
      for (int e = 0; e < 16; ++e) {
        const int kk = 32 * kc + e + hi * 16;
        af[kc].s[e] = s_ea[lm16][kk];
      }
    }
    v8f acc0 = {}, acc1 = {};           // two chains: halve WMMA RAW latency
    acc0 = __builtin_amdgcn_wmma_f32_16x16x32_bf16(false, af[0].v, false, bf[0].v, (short)0, acc0, false, false);
    acc1 = __builtin_amdgcn_wmma_f32_16x16x32_bf16(false, af[1].v, false, bf[1].v, (short)0, acc1, false, false);
    acc0 = __builtin_amdgcn_wmma_f32_16x16x32_bf16(false, af[2].v, false, bf[2].v, (short)0, acc0, false, false);
    acc1 = __builtin_amdgcn_wmma_f32_16x16x32_bf16(false, af[3].v, false, bf[3].v, (short)0, acc1, false, false);
    const v8f acc = acc0 + acc1;

    // ---- write back: alpha' = emission + m + log(sum)  (sum >= ~0.2, safe) ----
    #pragma unroll
    for (int r = 0; r < 8; ++r) {
      const int m = r + hi * 8;                 // C/D layout: lanes>=16 hold M+8
      s_alpha[m][n] = em[r] + s_rowmax[m] + fast_log(acc[r]);
    }
    if (t < tmax) {  // pull next step's emissions toward the caches
      __builtin_prefetch(&inputs[((size_t)(b0 + hi * 8) * LL + t + 1) * TT + n], 0, 0);
    }
    __syncthreads();
  }

  // ---- final logsumexp for rows whose snapshot is the last processed step ----
  float lm = -3.0e38f;
  #pragma unroll
  for (int i = 0; i < 8; ++i) lm = fmaxf(lm, s_alpha[row][sub + 16 * i]);
  #pragma unroll
  for (int off = 8; off > 0; off >>= 1) lm = fmaxf(lm, __shfl_xor(lm, off, 16));
  float ls = 0.f;
  #pragma unroll
  for (int i = 0; i < 8; ++i) ls += fast_exp(s_alpha[row][sub + 16 * i] - lm);
  #pragma unroll
  for (int off = 8; off > 0; off >>= 1) ls += __shfl_xor(ls, off, 16);
  if (sub == 0 && s_snap[row] == tmax) s_lognorm[row] = lm + fast_log(ls);
  __syncthreads();

  if (tid < 16) out[b0 + tid] = score_ws[b0 + tid] - s_lognorm[tid];
}

// ---------------------------------------------------------------------------
extern "C" void kernel_launch(void* const* d_in, const int* in_sizes, int n_in,
                              void* d_out, int out_size, void* d_ws, size_t ws_size,
                              hipStream_t stream) {
  const float* inputs = (const float*)d_in[0];   // (256,1024,128) f32
  const int*   tags   = (const int*)d_in[1];     // (256,1024) int32
  const int*   lens   = (const int*)d_in[2];     // (256,) int32
  const float* trans  = (const float*)d_in[3];   // (128,128) f32
  float* out = (float*)d_out;                    // (256,) f32
  float* ws  = (float*)d_ws;                     // scores scratch (256 f32)

  crf_score_kernel<<<BB, 256, 0, stream>>>(inputs, tags, lens, trans, ws);
  crf_forward_kernel<<<BB / 16, 256, 0, stream>>>(inputs, lens, trans, ws, out);
}
